// MultiheadSelfAttention_31937376813367
// MI455X (gfx1250) — compile-verified, matrix operand order checked by reading, not run on hardware
//
#include <hip/hip_runtime.h>
#include <hip/hip_bf16.h>
#include <math.h>

// Problem constants (from reference)
#define D_MODEL 768
#define NHEAD   12
#define DHEAD   64
#define BATCH   2
#define SEQ     4096
#define ROWS    (BATCH*SEQ)     // 8192
#define EQKV    (3*D_MODEL)     // 2304

typedef __bf16 bf16;
typedef __attribute__((ext_vector_type(16))) __bf16 v16bf;
typedef __attribute__((ext_vector_type(8)))  __bf16 v8bf;
typedef __attribute__((ext_vector_type(8)))  float  v8f;

union V16U { v16bf v; v8bf h[2]; };

// ---------------------------------------------------------------------------
// Fragment loader following CDNA5 ISA §7.12.2 16-bit A-matrix layout:
//   lanes 0-15 : row M = lane,    K = {0..7, 16..23} (+k0)
//   lanes 16-31: row M = lane-16, K = {8..15, 24..31} (+k0)
// Same pattern serves as the B-operand loader whenever B's columns are
// K-contiguous in memory (true for every operand here by construction).
// ---------------------------------------------------------------------------
__device__ __forceinline__ v16bf load_frag(const bf16* base, int row0, int ld, int k0) {
    int lane = threadIdx.x & 31;
    int half = lane >> 4;
    int r    = lane & 15;
    const bf16* p = base + (size_t)(row0 + r) * (size_t)ld + k0 + half * 8;
    V16U u;
    u.h[0] = *(const v8bf*)(p);        // K = k0 + half*8 + {0..7}
    u.h[1] = *(const v8bf*)(p + 16);   // K = k0 + 16 + half*8 + {0..7}
    return u.v;
}

__device__ __forceinline__ v8f wmma_bf16(v16bf a, v16bf b, v8f c) {
    // v_wmma_f32_16x16x32_bf16  (D = A x B + C, f32 accumulate)
    return __builtin_amdgcn_wmma_f32_16x16x32_bf16(
        false, a, false, b, (short)0, c, false, false);
}

// ---------------------------------------------------------------------------
// Kernel 0: fp32 -> bf16 conversions (x, and weights as [Wq;Wk;Wv] + Wo)
// ---------------------------------------------------------------------------
__global__ void cvt_kernel(const float* __restrict__ x,
                           const float* __restrict__ wq,
                           const float* __restrict__ wk,
                           const float* __restrict__ wv,
                           const float* __restrict__ wo,
                           bf16* __restrict__ xb,
                           bf16* __restrict__ wqkv,
                           bf16* __restrict__ wob) {
    size_t i  = (size_t)blockIdx.x * blockDim.x + threadIdx.x;
    size_t NX = (size_t)ROWS * D_MODEL;       // 6,291,456
    size_t NW = (size_t)D_MODEL * D_MODEL;    //   589,824
    if (i < NX) xb[i] = (bf16)x[i];
    if (i < NW) {
        wqkv[i]          = (bf16)wq[i];
        wqkv[NW + i]     = (bf16)wk[i];
        wqkv[2 * NW + i] = (bf16)wv[i];
        wob[i]           = (bf16)wo[i];
    }
}

// ---------------------------------------------------------------------------
// Register-blocked WMMA GEMM:  C[M,N] = A[M,K] * Bw[N,K]^T
// One wave per 32x64 macro-tile (MT=2 x NT=4 of 16x16 WMMA tiles):
//   per K=32 step: 6 fragment loads (6 KB) feed 8 WMMAs (131 KFLOP)
//   => ~21 FLOP/byte of VGPR fill vs 8 for a naive 16x16/wave tiling.
// ~140 VGPRs/lane (64 acc + 48 frag + addressing) keeps multi-wave occupancy.
// ---------------------------------------------------------------------------
#define GMT 2   // M tiles per wave
#define GNT 4   // N tiles per wave

template <typename OutT>
__global__ void gemm_kernel(const bf16* __restrict__ A,
                            const bf16* __restrict__ Bw,
                            OutT* __restrict__ C,
                            int M, int N, int K) {
    int macro_n = N / (16 * GNT);
    int id = blockIdx.x;
    int tm = (id / macro_n) * (16 * GMT);
    int tn = (id % macro_n) * (16 * GNT);

    v8f acc[GMT][GNT] = {};
    for (int k0 = 0; k0 < K; k0 += 32) {
        v16bf a[GMT], b[GNT];
#pragma unroll
        for (int mi = 0; mi < GMT; ++mi)
            a[mi] = load_frag(A, tm + 16 * mi, K, k0);
#pragma unroll
        for (int ni = 0; ni < GNT; ++ni)
            b[ni] = load_frag(Bw, tn + 16 * ni, K, k0);
        // prefetch next K step into cache ahead of the load clauses
        if (k0 + 32 < K) {
            int lane = threadIdx.x & 31, r = lane & 15;
            __builtin_prefetch(A  + (size_t)(tm + r) * K + k0 + 32, 0, 3);
            __builtin_prefetch(Bw + (size_t)(tn + r) * K + k0 + 32, 0, 3);
        }
#pragma unroll
        for (int mi = 0; mi < GMT; ++mi)
#pragma unroll
            for (int ni = 0; ni < GNT; ++ni)
                acc[mi][ni] = wmma_bf16(a[mi], b[ni], acc[mi][ni]);
    }

    int lane = threadIdx.x & 31, half = lane >> 4, r = lane & 15;
#pragma unroll
    for (int mi = 0; mi < GMT; ++mi)
#pragma unroll
        for (int ni = 0; ni < GNT; ++ni)
#pragma unroll
            for (int i = 0; i < 8; ++i)
                C[(size_t)(tm + 16 * mi + i + 8 * half) * (size_t)N + tn + 16 * ni + r]
                    = (OutT)acc[mi][ni][i];
}

// ---------------------------------------------------------------------------
// Kernel 2: RoPE + head reshape.
//   qkv [B,S,2304] bf16  ->  qr,kr [B,H,S,64] bf16 (roped), vt [B,H,64,S] bf16
// One wave per (b,h,s); lane = rotation pair index (d_head/2 = 32 pairs).
// ---------------------------------------------------------------------------
__global__ void rope_kernel(const bf16* __restrict__ qkv,
                            const int*  __restrict__ pos,
                            bf16* __restrict__ qr,
                            bf16* __restrict__ kr,
                            bf16* __restrict__ vt) {
    int idx = blockIdx.x;             // b*H*S + h*S + s
    int s  = idx % SEQ;
    int bh = idx / SEQ;
    int h  = bh % NHEAD;
    int b  = bh / NHEAD;
    int lane = threadIdx.x & 31;

    const bf16* row = qkv + (size_t)(b * SEQ + s) * EQKV;
    float p   = (float)pos[s];
    // inv_freq = 10000^(-lane/32)
    float inv = __expf(-((float)lane * (1.0f / 32.0f)) * 9.210340371976184f);
    float ang = p * inv;
    float sn, cs;
    __sincosf(ang, &sn, &cs);

    float q1 = (float)row[h * 64 + 2 * lane];
    float q2 = (float)row[h * 64 + 2 * lane + 1];
    float k1 = (float)row[D_MODEL + h * 64 + 2 * lane];
    float k2 = (float)row[D_MODEL + h * 64 + 2 * lane + 1];
    bf16  v1 = row[2 * D_MODEL + h * 64 + 2 * lane];
    bf16  v2 = row[2 * D_MODEL + h * 64 + 2 * lane + 1];

    size_t hb = (size_t)(b * NHEAD + h) * SEQ;
    bf16* qo = qr + (hb + s) * 64;
    qo[2 * lane]     = (bf16)(q1 * cs - q2 * sn);
    qo[2 * lane + 1] = (bf16)(q1 * sn + q2 * cs);
    bf16* ko = kr + (hb + s) * 64;
    ko[2 * lane]     = (bf16)(k1 * cs - k2 * sn);
    ko[2 * lane + 1] = (bf16)(k1 * sn + k2 * cs);
    // V transposed: vt[b,h,d,s]  (makes P*V B-operand K-contiguous)
    bf16* vo = vt + (size_t)(b * NHEAD + h) * (size_t)DHEAD * SEQ;
    vo[(size_t)(2 * lane)     * SEQ + s] = v1;
    vo[(size_t)(2 * lane + 1) * SEQ + s] = v2;
}

// ---------------------------------------------------------------------------
// Kernel 3: causal flash attention. One wave per 16-row Q tile.
//   S-tile: Q(16x64) * K^T via 2 WMMAs per 16-col block; KV tiles of 32.
//   Online softmax: row stats live per-lane (C layout: lanes 0-15 own rows
//   0-7, lanes 16-31 own rows 8-15), row reductions via 16-lane shfl_xor.
//   P (16x32) bounces through LDS to convert C-layout -> A-layout.
//   O accumulated as 4 C-tiles over d_head.
// ---------------------------------------------------------------------------
__global__ void attn_kernel(const bf16* __restrict__ qr,
                            const bf16* __restrict__ kr,
                            const bf16* __restrict__ vt,
                            bf16* __restrict__ ao) {
    __shared__ bf16 Pbuf[16 * 32];

    int qtile = blockIdx.x;           // 0..SEQ/16-1
    int h = blockIdx.y;
    int b = blockIdx.z;
    int lane = threadIdx.x & 31, half = lane >> 4, r = lane & 15;

    size_t hb = (size_t)(b * NHEAD + h) * SEQ;
    const bf16* qbase = qr + hb * 64;
    const bf16* kbase = kr + hb * 64;
    const bf16* vbase = vt + (size_t)(b * NHEAD + h) * (size_t)DHEAD * SEQ;
    int q0 = qtile * 16;

    v16bf qa0 = load_frag(qbase, q0, DHEAD, 0);
    v16bf qa1 = load_frag(qbase, q0, DHEAD, 32);

    float m[8], l[8];
    v8f o[4] = {};
#pragma unroll
    for (int i = 0; i < 8; ++i) { m[i] = -1e30f; l[i] = 0.0f; }

    const float scale = 0.125f;       // 1/sqrt(64)
    int kend = q0 + 15;               // causal bound

    for (int k0 = 0; k0 <= kend; k0 += 32) {
        // prefetch next KV tile while this one computes
        if (k0 + 32 <= kend)
            __builtin_prefetch(kbase + (size_t)(k0 + 32 + r) * DHEAD, 0, 3);

        // ---- scores for kv columns [k0, k0+32) ----
        v8f s0 = {}, s1 = {};
        {
            v16bf kb0a = load_frag(kbase, k0,      DHEAD, 0);
            v16bf kb0b = load_frag(kbase, k0,      DHEAD, 32);
            s0 = wmma_bf16(qa0, kb0a, s0);
            s0 = wmma_bf16(qa1, kb0b, s0);
            v16bf kb1a = load_frag(kbase, k0 + 16, DHEAD, 0);
            v16bf kb1b = load_frag(kbase, k0 + 16, DHEAD, 32);
            s1 = wmma_bf16(qa0, kb1a, s1);
            s1 = wmma_bf16(qa1, kb1b, s1);
        }
        // ---- causal mask + scale ----
        int qrow_base = q0 + 8 * half;
        int col0 = k0 + r, col1 = k0 + 16 + r;
        float p0[8], p1[8];
#pragma unroll
        for (int i = 0; i < 8; ++i) {
            int qrow = qrow_base + i;
            p0[i] = (col0 <= qrow) ? s0[i] * scale : -1e30f;
            p1[i] = (col1 <= qrow) ? s1[i] * scale : -1e30f;
        }
        // ---- online softmax update, store P to LDS as bf16 ----
#pragma unroll
        for (int i = 0; i < 8; ++i) {
            float tm = fmaxf(p0[i], p1[i]);
#pragma unroll
            for (int d = 1; d < 16; d <<= 1)
                tm = fmaxf(tm, __shfl_xor(tm, d, 32));
            float mn   = fmaxf(m[i], tm);
            float corr = __expf(m[i] - mn);
            float e0   = __expf(p0[i] - mn);
            float e1   = __expf(p1[i] - mn);
            float rs   = e0 + e1;
#pragma unroll
            for (int d = 1; d < 16; d <<= 1)
                rs += __shfl_xor(rs, d, 32);
            l[i] = l[i] * corr + rs;
            m[i] = mn;
#pragma unroll
            for (int t = 0; t < 4; ++t) o[t][i] *= corr;
            Pbuf[(i + 8 * half) * 32 + r]      = (bf16)e0;
            Pbuf[(i + 8 * half) * 32 + 16 + r] = (bf16)e1;
        }
        // ---- O += P * V  (P via LDS A-layout reload; V^T is K-contiguous) ----
        v16bf pa = load_frag(Pbuf, 0, 32, 0);
#pragma unroll
        for (int t = 0; t < 4; ++t) {
            v16bf vb = load_frag(vbase, t * 16, SEQ, k0);
            o[t] = wmma_bf16(pa, vb, o[t]);
        }
    }

    // ---- normalize and write: ao[b, s, h*64 + d] ----
#pragma unroll
    for (int i = 0; i < 8; ++i) {
        float inv_l = 1.0f / l[i];
        size_t srow = (size_t)(b * SEQ + q0 + i + 8 * half);
#pragma unroll
        for (int t = 0; t < 4; ++t)
            ao[srow * D_MODEL + h * 64 + t * 16 + r] = (bf16)(o[t][i] * inv_l);
    }
}

// ---------------------------------------------------------------------------
// Host-side orchestration
// ---------------------------------------------------------------------------
extern "C" void kernel_launch(void* const* d_in, const int* in_sizes, int n_in,
                              void* d_out, int out_size, void* d_ws, size_t ws_size,
                              hipStream_t stream) {
    const float* x  = (const float*)d_in[0];
    const float* Wq = (const float*)d_in[1];
    const float* Wk = (const float*)d_in[2];
    const float* Wv = (const float*)d_in[3];
    const float* Wo = (const float*)d_in[4];
    const int* tpos = (const int*)d_in[5];
    float* out = (float*)d_out;

    // Workspace layout (256B aligned).  ao aliases xb (xb dead after QKV GEMM).
    char* ws = (char*)d_ws;
    size_t off = 0;
    auto take = [&](size_t bytes) { size_t o = off; off = (off + bytes + 255) & ~(size_t)255; return o; };

    const size_t SZ_XB   = (size_t)ROWS * D_MODEL * sizeof(bf16);      // 12.6 MB
    const size_t SZ_WQKV = (size_t)EQKV * D_MODEL * sizeof(bf16);      //  3.5 MB
    const size_t SZ_WOB  = (size_t)D_MODEL * D_MODEL * sizeof(bf16);   //  1.2 MB
    const size_t SZ_QKV  = (size_t)ROWS * EQKV * sizeof(bf16);         // 37.7 MB
    const size_t SZ_HEAD = (size_t)BATCH * NHEAD * SEQ * DHEAD * sizeof(bf16); // 12.6 MB

    size_t o_xb   = take(SZ_XB);       // also reused as ao
    size_t o_wqkv = take(SZ_WQKV);
    size_t o_wob  = take(SZ_WOB);
    size_t o_qkv  = take(SZ_QKV);
    size_t o_qr   = take(SZ_HEAD);
    size_t o_kr   = take(SZ_HEAD);
    size_t o_vt   = take(SZ_HEAD);
    (void)ws_size;

    bf16* xb   = (bf16*)(ws + o_xb);
    bf16* wqkv = (bf16*)(ws + o_wqkv);
    bf16* wob  = (bf16*)(ws + o_wob);
    bf16* qkv  = (bf16*)(ws + o_qkv);
    bf16* qr   = (bf16*)(ws + o_qr);
    bf16* kr   = (bf16*)(ws + o_kr);
    bf16* vt   = (bf16*)(ws + o_vt);
    bf16* ao   = xb;                   // alias: x(bf16) is dead before attention writes

    // 0) fp32 -> bf16
    {
        size_t n = (size_t)ROWS * D_MODEL;
        int blocks = (int)((n + 255) / 256);
        cvt_kernel<<<blocks, 256, 0, stream>>>(x, Wq, Wk, Wv, Wo, xb, wqkv, wob);
    }
    // 1) QKV projection: [8192,2304] = xb[8192,768] * Wqkv[2304,768]^T
    {
        int grid = (ROWS / (16 * GMT)) * (EQKV / (16 * GNT));   // 256*36 = 9216 waves
        gemm_kernel<bf16><<<grid, 32, 0, stream>>>(xb, wqkv, qkv, ROWS, EQKV, D_MODEL);
    }
    // 2) RoPE + head split (+ V transpose)
    {
        int grid = BATCH * NHEAD * SEQ;         // 98304 waves
        rope_kernel<<<grid, 32, 0, stream>>>(qkv, tpos, qr, kr, vt);
    }
    // 3) causal flash attention
    {
        dim3 grid(SEQ / 16, NHEAD, BATCH);      // 256 x 12 x 2
        attn_kernel<<<grid, 32, 0, stream>>>(qr, kr, vt, ao);
    }
    // 4) output projection: out[8192,768] = ao[8192,768] * Wo[768,768]^T (f32 out)
    {
        int grid = (ROWS / (16 * GMT)) * (D_MODEL / (16 * GNT)); // 256*12 = 3072 waves
        gemm_kernel<float><<<grid, 32, 0, stream>>>(ao, wob, out, ROWS, D_MODEL, D_MODEL);
    }
}
